// GraphV1EmbLarge_73864847556672
// MI455X (gfx1250) — matrix operand order repaired and checked by reading, not compiled, not run on hardware
//
#include <hip/hip_runtime.h>

// ---------------- problem constants ----------------
#define N_NODES     10000
#define E_EDGES     80000
#define EMB_DIM     128
#define IN_CH       64
#define IN_GNN      191
#define K1P         192          // 191 padded to multiple of 32
#define C1          2048
#define C2          1024
#define C3          1024
#define M_PAD       10112        // 79 * 128, multiple of 128 for block tiling
#define NUM_GRAPHS  16
#define NUM_CLASSES 2
#define SCAN_NB     ((N_NODES + 255) / 256)   // 40 scan blocks

typedef __attribute__((ext_vector_type(16))) __bf16 bf16x16;
typedef __attribute__((ext_vector_type(8)))  float  floatx8;

__device__ __forceinline__ unsigned short f2bf(float f) {
    unsigned int u = __float_as_uint(f);
    u += 0x7FFFu + ((u >> 16) & 1u);           // round-to-nearest-even
    return (unsigned short)(u >> 16);
}
__device__ __forceinline__ float bf2f(unsigned short b) {
    return __uint_as_float(((unsigned int)b) << 16);
}

// ---------------- feature build: x = [emb[type], other_feats, pad] -> bf16 ----------------
__global__ void build_x_kernel(const int* __restrict__ type_ids,
                               const float* __restrict__ other,
                               const float* __restrict__ emb,
                               unsigned short* __restrict__ Xbf) {
    long long idx = (long long)blockIdx.x * blockDim.x + threadIdx.x;
    int i = (int)(idx / K1P), c = (int)(idx % K1P);
    if (i >= N_NODES) return;
    float v = 0.f;
    if (c < EMB_DIM)      v = emb[(size_t)type_ids[i] * EMB_DIM + c];
    else if (c < IN_GNN)  v = other[(size_t)i * (IN_CH - 1) + (c - EMB_DIM)];
    Xbf[(size_t)i * K1P + c] = f2bf(v);
}

// ---------------- weight transpose + bf16 convert: W[K,N] -> Wt[N,Kpad] ----------------
__global__ void wt_bf16_kernel(const float* __restrict__ W, unsigned short* __restrict__ Wt,
                               int K, int Kpad, int Nout) {
    long long idx = (long long)blockIdx.x * blockDim.x + threadIdx.x;
    int o = (int)(idx / Kpad), k = (int)(idx % Kpad);
    if (o >= Nout) return;
    float v = (k < K) ? W[(size_t)k * Nout + o] : 0.f;
    Wt[(size_t)o * Kpad + k] = f2bf(v);
}

// ---------------- GCN normalization + CSR build ----------------
__global__ void deg_count_kernel(const int* __restrict__ dst, unsigned int* __restrict__ deg) {
    int e = blockIdx.x * blockDim.x + threadIdx.x;
    if (e >= E_EDGES) return;
    atomicAdd(&deg[dst[e]], 1u);
}
__global__ void dinv_kernel(const unsigned int* __restrict__ deg, float* __restrict__ dinv) {
    int i = blockIdx.x * blockDim.x + threadIdx.x;
    if (i >= N_NODES) return;
    dinv[i] = rsqrtf((float)(deg[i] + 1u));    // +1 = self loop
}

// phase A: per-block exclusive scan of deg, block totals to bsum
__global__ void scan_block_kernel(const unsigned int* __restrict__ deg,
                                  unsigned int* __restrict__ offs,
                                  unsigned int* __restrict__ bsum) {
    __shared__ unsigned int sh[256];
    const int t = threadIdx.x;
    const int i = blockIdx.x * 256 + t;
    unsigned int v = (i < N_NODES) ? deg[i] : 0u;
    sh[t] = v;
    __syncthreads();
    for (int s = 1; s < 256; s <<= 1) {
        unsigned int add = (t >= s) ? sh[t - s] : 0u;
        __syncthreads();
        sh[t] += add;
        __syncthreads();
    }
    if (i < N_NODES) offs[i] = sh[t] - v;       // exclusive within block
    if (t == 255)   bsum[blockIdx.x] = sh[255]; // block total
}
// phase B: serial exclusive scan of the 40 block sums (+ total at bsum[SCAN_NB])
__global__ void scan_bsum_kernel(unsigned int* __restrict__ bsum) {
    if (threadIdx.x == 0 && blockIdx.x == 0) {
        unsigned int acc = 0;
        for (int i = 0; i < SCAN_NB; ++i) { unsigned int t = bsum[i]; bsum[i] = acc; acc += t; }
        bsum[SCAN_NB] = acc;
    }
}
// phase C: add block offsets; offs[N] = total
__global__ void scan_add_kernel(unsigned int* __restrict__ offs, const unsigned int* __restrict__ bsum) {
    int i = blockIdx.x * 256 + threadIdx.x;
    if (i < N_NODES)       offs[i] += bsum[i >> 8];
    else if (i == N_NODES) offs[N_NODES] = bsum[SCAN_NB];
}
// counting-sort edges by dst into CSR
__global__ void csr_fill_kernel(const int* __restrict__ src, const int* __restrict__ dst,
                                const unsigned int* __restrict__ offs,
                                unsigned int* __restrict__ cursor, int* __restrict__ csr_src) {
    int e = blockIdx.x * blockDim.x + threadIdx.x;
    if (e >= E_EDGES) return;
    const int d = dst[e];
    const unsigned int pos = offs[d] + atomicAdd(&cursor[d], 1u);
    csr_src[pos] = src[e];
}

// ---------------- bf16 WMMA GEMM: Y[M_PAD,Nout](bf16) = X[M_PAD,K](bf16) @ Wt[Nout,K]^T ----
// block = 256 threads = 8 waves; wave w -> rows [blockIdx.y*128 + w*16, +16),
// cols [blockIdx.x*64, +64) as 4 x (16x16) accumulators. B tile staged in LDS.
__global__ __launch_bounds__(256, 2)
void gemm_bf16_wmma_kernel(const unsigned short* __restrict__ X,
                           const unsigned short* __restrict__ Wt,
                           unsigned short* __restrict__ Ybf,
                           int K, int Nout) {
    __shared__ unsigned short ldsB[4 * 32 * 16];   // 4 tiles x 32 lanes x 16 bf16 = 4KB

    const int tid    = threadIdx.x;
    const int wave   = tid >> 5;
    const int lane   = tid & 31;
    const int laneLo = lane & 15;
    const int half   = lane >> 4;

    const int mBase = (blockIdx.y * 8 + wave) * 16;
    const int nBase = blockIdx.x * 64;

    const unsigned short* arow = X + (size_t)(mBase + laneLo) * K;

    // cooperative B-fill mapping: tid -> (tile t, lane l, 8-elem half jh)
    const int ft = tid >> 6;
    const int fl = (tid >> 1) & 31;
    const int fj = tid & 1;
    const unsigned short* bsrc =
        Wt + (size_t)(nBase + ft * 16 + (fl & 15)) * K + ((fl >> 4) * 8 + fj * 16);
    unsigned short* bdst = &ldsB[((ft * 32 + fl) * 16) + fj * 8];

    floatx8 acc0 = {}, acc1 = {}, acc2 = {}, acc3 = {};

    for (int k0 = 0; k0 < K; k0 += 32) {
        // stage B tile (64 cols x 32 k) into LDS in fragment layout
        *(uint4*)bdst = *(const uint4*)(bsrc + k0);
        __syncthreads();

        // A fragment (ISA 16-bit A 16x32 layout)
        union { bf16x16 v; uint4 q[2]; } a;
        const int ka = k0 + half * 8;
        a.q[0] = *(const uint4*)(arow + ka);
        a.q[1] = *(const uint4*)(arow + ka + 16);

        bf16x16 b0 = *(const bf16x16*)&ldsB[(0 * 32 + lane) * 16];
        bf16x16 b1 = *(const bf16x16*)&ldsB[(1 * 32 + lane) * 16];
        bf16x16 b2 = *(const bf16x16*)&ldsB[(2 * 32 + lane) * 16];
        bf16x16 b3 = *(const bf16x16*)&ldsB[(3 * 32 + lane) * 16];

        acc0 = __builtin_amdgcn_wmma_f32_16x16x32_bf16(false, a.v, false, b0, (short)0, acc0, false, false);
        acc1 = __builtin_amdgcn_wmma_f32_16x16x32_bf16(false, a.v, false, b1, (short)0, acc1, false, false);
        acc2 = __builtin_amdgcn_wmma_f32_16x16x32_bf16(false, a.v, false, b2, (short)0, acc2, false, false);
        acc3 = __builtin_amdgcn_wmma_f32_16x16x32_bf16(false, a.v, false, b3, (short)0, acc3, false, false);
        __syncthreads();
    }

    // D layout: VGPR r -> (M = mBase + half*8 + r, N = nBase + t*16 + laneLo)
    #pragma unroll
    for (int r = 0; r < 8; ++r) {
        size_t row = (size_t)(mBase + half * 8 + r) * Nout + nBase + laneLo;
        Ybf[row +  0] = f2bf(acc0[r]);
        Ybf[row + 16] = f2bf(acc1[r]);
        Ybf[row + 32] = f2bf(acc2[r]);
        Ybf[row + 48] = f2bf(acc3[r]);
    }
}

// ------- fused CSR gather + self-loop + bias + relu -> bf16 and/or f32 (no atomics) -------
// agg[d] = relu( dinv[d] * sum_{s in N(d)} dinv[s]*h[s]  +  dinv[d]^2 * h[d]  +  bias )
__global__ void gather_combine_kernel(const unsigned int* __restrict__ offs,
                                      const int* __restrict__ csr_src,
                                      const float* __restrict__ dinv,
                                      const unsigned short* __restrict__ H,
                                      const float* __restrict__ bias, int Nout,
                                      unsigned short* __restrict__ outBf,
                                      float* __restrict__ outF) {
    const int chunks = Nout >> 3;
    long long idx = (long long)blockIdx.x * blockDim.x + threadIdx.x;
    const int d  = (int)(idx / chunks);          // warp-uniform (chunks >= 32)
    const int ch = (int)(idx % chunks) * 8;
    if (d >= N_NODES) return;

    float acc[8] = {0.f, 0.f, 0.f, 0.f, 0.f, 0.f, 0.f, 0.f};
    const unsigned int e0 = offs[d], e1 = offs[d + 1];
    for (unsigned int e = e0; e < e1; ++e) {
        const int s   = csr_src[e];
        const float w = dinv[s];
        union { uint4 q; unsigned short u[8]; } h;
        h.q = *(const uint4*)(H + (size_t)s * Nout + ch);
        #pragma unroll
        for (int j = 0; j < 8; ++j) acc[j] = fmaf(w, bf2f(h.u[j]), acc[j]);
    }

    const float dd = dinv[d];
    union { uint4 q; unsigned short u[8]; } hs;
    hs.q = *(const uint4*)(H + (size_t)d * Nout + ch);

    union { uint4 q; unsigned short u[8]; } ob;
    float of[8];
    #pragma unroll
    for (int j = 0; j < 8; ++j) {
        float v = fmaf(acc[j], dd, bf2f(hs.u[j]) * dd * dd) + bias[ch + j];
        v = fmaxf(v, 0.f);
        ob.u[j] = f2bf(v);
        of[j] = v;
    }
    if (outBf) *(uint4*)(outBf + (size_t)d * Nout + ch) = ob.q;
    if (outF) {
        float* p = outF + (size_t)d * Nout + ch;
        *(float4*)p       = make_float4(of[0], of[1], of[2], of[3]);
        *(float4*)(p + 4) = make_float4(of[4], of[5], of[6], of[7]);
    }
}

// ---------------- global mean pool ----------------
__global__ void pool_sum_kernel(const float* __restrict__ x, const int* __restrict__ batch,
                                float* __restrict__ sums) {
    long long idx = (long long)blockIdx.x * blockDim.x + threadIdx.x;
    int i = (int)(idx / C3), c = (int)(idx % C3);
    if (i >= N_NODES) return;
    atomicAdd(&sums[(size_t)batch[i] * C3 + c], x[(size_t)i * C3 + c]);
}
__global__ void pool_cnt_kernel(const int* __restrict__ batch, float* __restrict__ cnt) {
    int i = blockIdx.x * blockDim.x + threadIdx.x;
    if (i >= N_NODES) return;
    atomicAdd(&cnt[batch[i]], 1.f);
}
__global__ void pool_div_kernel(const float* __restrict__ sums, const float* __restrict__ cnt,
                                float* __restrict__ g) {
    int idx = blockIdx.x * blockDim.x + threadIdx.x;
    if (idx >= NUM_GRAPHS * C3) return;
    g[idx] = sums[idx] / fmaxf(cnt[idx / C3], 1.f);
}

// ---------------- tiny dense head (16 rows) ----------------
__global__ void dense_kernel(const float* __restrict__ in, const float* __restrict__ W,
                             const float* __restrict__ b, float* __restrict__ out,
                             int K, int Nout, int relu) {
    int idx = blockIdx.x * blockDim.x + threadIdx.x;
    int gi = idx / Nout, n = idx % Nout;
    if (gi >= NUM_GRAPHS) return;
    float s = b[n];
    for (int k = 0; k < K; ++k) s += in[(size_t)gi * K + k] * W[(size_t)k * Nout + n];
    out[(size_t)gi * Nout + n] = relu ? fmaxf(s, 0.f) : s;
}

// ------------------------------------------------------------------
extern "C" void kernel_launch(void* const* d_in, const int* in_sizes, int n_in,
                              void* d_out, int out_size, void* d_ws, size_t ws_size,
                              hipStream_t stream) {
    const int*   type_ids = (const int*)d_in[0];
    const float* other    = (const float*)d_in[1];
    const int*   eidx     = (const int*)d_in[2];     // [2,E] row-major: src then dst
    const int*   batch    = (const int*)d_in[3];
    const float* emb      = (const float*)d_in[4];
    const float* W1 = (const float*)d_in[5];  const float* b1 = (const float*)d_in[6];
    const float* W2 = (const float*)d_in[7];  const float* b2 = (const float*)d_in[8];
    const float* W3 = (const float*)d_in[9];  const float* b3 = (const float*)d_in[10];
    const float* h1w = (const float*)d_in[11]; const float* h1b = (const float*)d_in[12];
    const float* h2w = (const float*)d_in[13]; const float* h2b = (const float*)d_in[14];
    const float* h3w = (const float*)d_in[15]; const float* h3b = (const float*)d_in[16];
    const float* ow  = (const float*)d_in[17]; const float* ob  = (const float*)d_in[18];
    const int* esrc = eidx;
    const int* edst = eidx + E_EDGES;

    // -------- workspace carve --------
    size_t off = 0;
    auto carve = [&](size_t bytes) -> char* {
        char* p = (char*)d_ws + off;
        off += (bytes + 255) & ~(size_t)255;
        return p;
    };
    unsigned short* Xbf0 = (unsigned short*)carve((size_t)M_PAD * K1P * 2);
    unsigned short* W1t  = (unsigned short*)carve((size_t)C1 * K1P * 2);
    unsigned short* W2t  = (unsigned short*)carve((size_t)C2 * C1 * 2);
    unsigned short* W3t  = (unsigned short*)carve((size_t)C3 * C2 * 2);
    unsigned short* Hbf  = (unsigned short*)carve((size_t)M_PAD * C1 * 2);
    unsigned short* XA   = (unsigned short*)carve((size_t)M_PAD * C1 * 2); // layer2 in; reused as f32 layer3 out
    unsigned short* XB   = (unsigned short*)carve((size_t)M_PAD * C2 * 2); // layer3 in
    unsigned int*   deg  = (unsigned int*)carve((size_t)N_NODES * 4);
    float*          dinv = (float*)carve((size_t)N_NODES * 4);
    unsigned int*   offs = (unsigned int*)carve((size_t)(N_NODES + 1) * 4);
    unsigned int*   bsum = (unsigned int*)carve((size_t)(SCAN_NB + 1) * 4);
    unsigned int*   curs = (unsigned int*)carve((size_t)N_NODES * 4);
    int*            csrs = (int*)carve((size_t)E_EDGES * 4);
    float*          sums = (float*)carve((size_t)NUM_GRAPHS * C3 * 4);
    float*          cnt  = (float*)carve((size_t)NUM_GRAPHS * 4);
    float*          g0   = (float*)carve((size_t)NUM_GRAPHS * 1024 * 4);
    float*          g1   = (float*)carve((size_t)NUM_GRAPHS * 1024 * 4);
    float*          g2   = (float*)carve((size_t)NUM_GRAPHS * 1024 * 4);
    float*          g3   = (float*)carve((size_t)NUM_GRAPHS * 512 * 4);
    float*          Xf3  = (float*)XA;

    const int B = 256;
    auto blocks = [&](long long n) { return (unsigned)((n + B - 1) / B); };

    // -------- features + weights --------
    hipMemsetAsync(Xbf0, 0, (size_t)M_PAD * K1P * 2, stream);
    build_x_kernel<<<blocks((long long)N_NODES * K1P), B, 0, stream>>>(type_ids, other, emb, Xbf0);
    wt_bf16_kernel<<<blocks((long long)C1 * K1P), B, 0, stream>>>(W1, W1t, IN_GNN, K1P, C1);
    wt_bf16_kernel<<<blocks((long long)C2 * C1), B, 0, stream>>>(W2, W2t, C1, C1, C2);
    wt_bf16_kernel<<<blocks((long long)C3 * C2), B, 0, stream>>>(W3, W3t, C2, C2, C3);

    // -------- gcn norm + CSR (by dst) --------
    hipMemsetAsync(deg, 0, (size_t)N_NODES * 4, stream);
    deg_count_kernel<<<blocks(E_EDGES), B, 0, stream>>>(edst, deg);
    dinv_kernel<<<blocks(N_NODES), B, 0, stream>>>(deg, dinv);
    scan_block_kernel<<<SCAN_NB, 256, 0, stream>>>(deg, offs, bsum);
    scan_bsum_kernel<<<1, 32, 0, stream>>>(bsum);
    scan_add_kernel<<<blocks(N_NODES + 1), 256, 0, stream>>>(offs, bsum);
    hipMemsetAsync(curs, 0, (size_t)N_NODES * 4, stream);
    csr_fill_kernel<<<blocks(E_EDGES), B, 0, stream>>>(esrc, edst, offs, curs, csrs);

    // -------- layer 1: [M,192] x [192,2048] --------
    gemm_bf16_wmma_kernel<<<dim3(C1 / 64, M_PAD / 128), B, 0, stream>>>(Xbf0, W1t, Hbf, K1P, C1);
    hipMemsetAsync(XA, 0, (size_t)M_PAD * C1 * 2, stream);
    gather_combine_kernel<<<blocks((long long)N_NODES * (C1 / 8)), B, 0, stream>>>(
        offs, csrs, dinv, Hbf, b1, C1, XA, nullptr);

    // -------- layer 2: [M,2048] x [2048,1024] --------
    gemm_bf16_wmma_kernel<<<dim3(C2 / 64, M_PAD / 128), B, 0, stream>>>(XA, W2t, Hbf, C1, C2);
    hipMemsetAsync(XB, 0, (size_t)M_PAD * C2 * 2, stream);
    gather_combine_kernel<<<blocks((long long)N_NODES * (C2 / 8)), B, 0, stream>>>(
        offs, csrs, dinv, Hbf, b2, C2, XB, nullptr);

    // -------- layer 3: [M,1024] x [1024,1024] --------
    gemm_bf16_wmma_kernel<<<dim3(C3 / 64, M_PAD / 128), B, 0, stream>>>(XB, W3t, Hbf, C2, C3);
    gather_combine_kernel<<<blocks((long long)N_NODES * (C3 / 8)), B, 0, stream>>>(
        offs, csrs, dinv, Hbf, b3, C3, nullptr, Xf3);

    // -------- pool + head --------
    hipMemsetAsync(sums, 0, (size_t)NUM_GRAPHS * C3 * 4, stream);
    hipMemsetAsync(cnt, 0, (size_t)NUM_GRAPHS * 4, stream);
    pool_sum_kernel<<<blocks((long long)N_NODES * C3), B, 0, stream>>>(Xf3, batch, sums);
    pool_cnt_kernel<<<blocks(N_NODES), B, 0, stream>>>(batch, cnt);
    pool_div_kernel<<<blocks(NUM_GRAPHS * C3), B, 0, stream>>>(sums, cnt, g0);

    dense_kernel<<<blocks(NUM_GRAPHS * 1024), B, 0, stream>>>(g0, h1w, h1b, g1, 1024, 1024, 1);
    dense_kernel<<<blocks(NUM_GRAPHS * 1024), B, 0, stream>>>(g1, h2w, h2b, g2, 1024, 1024, 1);
    dense_kernel<<<blocks(NUM_GRAPHS * 512), B, 0, stream>>>(g2, h3w, h3b, g3, 1024, 512, 1);
    dense_kernel<<<blocks(NUM_GRAPHS * NUM_CLASSES), B, 0, stream>>>(g3, ow, ob, (float*)d_out, 512, NUM_CLASSES, 0);
}
